// MultiHeadAttentionBlock_1297080124128
// MI455X (gfx1250) — matrix operand order repaired and checked by reading, not compiled, not run on hardware
//
#include <hip/hip_runtime.h>

// ---------------------------------------------------------------------------
// MHA block for MI455X (gfx1250, wave32). All matmuls via
// v_wmma_f32_16x16x32_bf16. K/V tiles DMA'd by the Tensor Data Mover
// (tensor_load_to_lds, TENSORcnt), GEMM weight tiles via async LDS loads
// (ASYNCcnt). Row-sum of softmax P computed with a WMMA against ones.
// ---------------------------------------------------------------------------

typedef __attribute__((ext_vector_type(16))) __bf16 bf16x16;
typedef __attribute__((ext_vector_type(8)))  float  f32x8;
typedef __attribute__((ext_vector_type(4)))  unsigned int u32x4;
typedef __attribute__((ext_vector_type(8)))  int i32x8;
typedef __attribute__((ext_vector_type(4)))  int i32x4;

union FragB16 { bf16x16 v; uint4 q[2]; };
union AccF    { f32x8 v; float f[8]; };

constexpr int Bn = 2, Sq = 4096, Dm = 768, Hh = 12, DKd = 64;

__device__ __forceinline__ unsigned short f2bf(float f) {
  union { float f; unsigned u; } c; c.f = f;
  unsigned u = c.u;                          // round-to-nearest-even f32->bf16
  return (unsigned short)((u + 0x7FFFu + ((u >> 16) & 1u)) >> 16);
}

__device__ __forceinline__ f32x8 wmma_bf16(bf16x16 a, bf16x16 b, f32x8 c) {
  return __builtin_amdgcn_wmma_f32_16x16x32_bf16(false, a, false, b,
                                                 (short)0, c, false, false);
}

// ---- TDM 2-D tile load: global[dim1][dim0] -> LDS tile [tile1][tile0] ------
// D# per CDNA5 ISA ch.8: group0 = {count/lds_addr/global_addr/type},
// group1 = {data_size=2B, dims, tile dims, stride0}. Groups 2/3 zero (2-D).
__device__ __forceinline__ void tdm_load_2d(unsigned lds_addr, const void* gptr,
                                            unsigned dim0, unsigned dim1,
                                            unsigned long long stride0,
                                            unsigned tile0, unsigned tile1) {
  unsigned long long ga = (unsigned long long)(uintptr_t)gptr;
  u32x4 g0 = {1u,                                   // count=1, user mode
              lds_addr,                             // LDS byte address
              (unsigned)(ga & 0xFFFFFFFFull),       // global_addr[31:0]
              (unsigned)((ga >> 32) & 0x01FFFFFFull) | (2u << 30)}; // type=2
  i32x8 g1 = {(int)(1u << 16),                      // data_size = 2 bytes
              (int)((dim0 & 0xFFFFu) << 16),        // tensor_dim0[15:0]
              (int)((dim0 >> 16) | ((dim1 & 0xFFFFu) << 16)),
              (int)((dim1 >> 16) | (tile0 << 16)),  // tile_dim0
              (int)tile1,                           // tile_dim1 (tile_dim2=0)
              (int)(stride0 & 0xFFFFFFFFull),       // tensor_dim0_stride
              (int)((stride0 >> 32) & 0xFFFFull),
              0};
  i32x4 gz = {0, 0, 0, 0};
#if defined(__clang_major__) && (__clang_major__ >= 23)
  i32x8 gz8 = {0, 0, 0, 0, 0, 0, 0, 0};
  __builtin_amdgcn_tensor_load_to_lds(g0, g1, gz, gz, gz8, 0);
#else
  __builtin_amdgcn_tensor_load_to_lds(g0, g1, gz, gz, 0);
#endif
}

__device__ __forceinline__ void async_lds_b128(unsigned lds_addr,
                                               const void* gptr) {
  unsigned long long ga = (unsigned long long)(uintptr_t)gptr;
  asm volatile("global_load_async_to_lds_b128 %0, %1, off"
               :: "v"(lds_addr), "v"(ga) : "memory");
}
__device__ __forceinline__ void wait_asynccnt0() {
  asm volatile("s_wait_asynccnt 0x0" ::: "memory");
}

// ---- weight transpose + f32->bf16: Wt[n*768+k] = bf16(W[k*768+n]) ----------
__global__ __launch_bounds__(256) void wtrans_kernel(
    const float* __restrict__ W, unsigned short* __restrict__ Wt) {
  int idx = blockIdx.x * 256 + threadIdx.x;
  int n = idx / Dm, k = idx - n * Dm;
  Wt[idx] = f2bf(W[(size_t)k * Dm + n]);
}

// ---- generic GEMM: Out = A[M,768] * W[768,768] + bias ----------------------
// Wt bf16 N-major. Block tile 128(M) x 64(N), 8 waves, wave = 16x64 strip.
// OUT_MODE: 0 = bf16 [B,H,S,DK]; 1 = bf16 [B,H,DK,S]; 2 = f32 row-major.
template <bool A_BF16, int OUT_MODE>
__global__ __launch_bounds__(256) void gemm_kernel(
    const void* __restrict__ Ap, const unsigned short* __restrict__ Wt,
    const float* __restrict__ bias, void* __restrict__ Outp) {
  constexpr int K = Dm, N = Dm;
  __shared__ alignas(16) unsigned short As[128][32];   // [m][k]
  __shared__ alignas(16) unsigned short Bs[64][32];    // [n][k]

  const int m0 = blockIdx.y * 128;
  const int n0 = blockIdx.x * 64;
  const int t  = threadIdx.x;
  const int wv = t >> 5, lane = t & 31;
  const int hl = lane >> 4, ln = lane & 15;

  AccF acc[4];
#pragma unroll
  for (int j = 0; j < 4; ++j)
#pragma unroll
    for (int i = 0; i < 8; ++i) acc[j].f[i] = 0.f;

  for (int kk = 0; kk < K; kk += 32) {
    {   // stage A tile 128x32
      int row = t >> 1, hf = t & 1;
      if (A_BF16) {
        async_lds_b128((unsigned)(uintptr_t)&As[row][hf * 16],
                       (const unsigned short*)Ap + (size_t)(m0 + row) * K + kk + hf * 16);
        async_lds_b128((unsigned)(uintptr_t)&As[row][hf * 16 + 8],
                       (const unsigned short*)Ap + (size_t)(m0 + row) * K + kk + hf * 16 + 8);
      } else {
        const float4* g4 = reinterpret_cast<const float4*>(
            (const float*)Ap + (size_t)(m0 + row) * K + kk + hf * 16);
        float4 a0 = g4[0], a1 = g4[1], a2 = g4[2], a3 = g4[3];
        float fs[16] = {a0.x, a0.y, a0.z, a0.w, a1.x, a1.y, a1.z, a1.w,
                        a2.x, a2.y, a2.z, a2.w, a3.x, a3.y, a3.z, a3.w};
        union { unsigned short h[16]; uint4 q[2]; } pk;
#pragma unroll
        for (int j = 0; j < 16; ++j) pk.h[j] = f2bf(fs[j]);
        *reinterpret_cast<uint4*>(&As[row][hf * 16])     = pk.q[0];
        *reinterpret_cast<uint4*>(&As[row][hf * 16 + 8]) = pk.q[1];
      }
    }
    {   // stage W tile 64(n) x 32(k) via async LDS DMA
      int n = t >> 2, sg = t & 3;
      async_lds_b128((unsigned)(uintptr_t)&Bs[n][sg * 8],
                     Wt + (size_t)(n0 + n) * K + kk + sg * 8);
    }
    wait_asynccnt0();
    __syncthreads();

    FragB16 afrag;
    const uint4* arow = reinterpret_cast<const uint4*>(&As[wv * 16 + ln][0]);
    afrag.q[0] = arow[hl];
    afrag.q[1] = arow[2 + hl];
#pragma unroll
    for (int t4 = 0; t4 < 4; ++t4) {
      FragB16 bfrag;
      const uint4* brow = reinterpret_cast<const uint4*>(&Bs[t4 * 16 + ln][0]);
      bfrag.q[0] = brow[hl];
      bfrag.q[1] = brow[2 + hl];
      acc[t4].v = wmma_bf16(afrag.v, bfrag.v, acc[t4].v);
    }
    __syncthreads();
  }

#pragma unroll
  for (int t4 = 0; t4 < 4; ++t4)
#pragma unroll
    for (int i = 0; i < 8; ++i) {
      int m = m0 + wv * 16 + i + 8 * hl;
      int n = n0 + t4 * 16 + ln;
      float val = acc[t4].f[i] + bias[n];
      if (OUT_MODE == 2) {
        ((float*)Outp)[(size_t)m * N + n] = val;
      } else {
        int b = m >> 12, s = m & (Sq - 1);
        int h = n >> 6, dk = n & (DKd - 1);
        unsigned short* o = (unsigned short*)Outp;
        if (OUT_MODE == 0)
          o[((size_t)(b * Hh + h) * Sq + s) * DKd + dk] = f2bf(val);
        else
          o[((size_t)(b * Hh + h) * DKd + dk) * Sq + s] = f2bf(val);
      }
    }
}

// ---- flash attention: per (b,h), 128 q rows/block, 32 keys/step, TDM -------
__global__ __launch_bounds__(256) void attn_kernel(
    const unsigned short* __restrict__ Qb, const unsigned short* __restrict__ Kb,
    const unsigned short* __restrict__ Vt, const int* __restrict__ mask,
    unsigned short* __restrict__ O) {
  __shared__ alignas(16) unsigned short Kt[2][32][64];   // [buf][key][dk]
  __shared__ alignas(16) unsigned short Vs[2][64][32];   // [buf][dk][key]
  __shared__ alignas(16) unsigned short Pb[8][16][32];   // per-wave P staging

  const int qt = blockIdx.x;
  const int bh = blockIdx.y;
  const int b  = bh / Hh;
  const int t  = threadIdx.x;
  const int wv = t >> 5, lane = t & 31;
  const int hl = lane >> 4, ln = lane & 15;

  const int qrow = qt * 128 + wv * 16 + ln;
  const uint4* qg = reinterpret_cast<const uint4*>(
      Qb + ((size_t)bh * Sq + qrow) * DKd);
  FragB16 qa0, qa1;
  qa0.q[0] = qg[2 * hl];     qa0.q[1] = qg[2 * hl + 1];
  qa1.q[0] = qg[4 + 2 * hl]; qa1.q[1] = qg[4 + 2 * hl + 1];

  FragB16 ones;          // bf16 1.0 broadcast: rowsum-of-P via WMMA
  ones.q[0] = make_uint4(0x3F803F80u, 0x3F803F80u, 0x3F803F80u, 0x3F803F80u);
  ones.q[1] = ones.q[0];

  AccF acc[4];
  float lrow[8];
  float Mrun = -1e30f;
#pragma unroll
  for (int i = 0; i < 8; ++i) lrow[i] = 0.f;
#pragma unroll
  for (int j = 0; j < 4; ++j)
#pragma unroll
    for (int i = 0; i < 8; ++i) acc[j].f[i] = 0.f;

  const unsigned short* Kg = Kb + (size_t)bh * Sq * DKd;
  const unsigned short* Vg = Vt + (size_t)bh * DKd * Sq;
  const int* mg = mask + (size_t)b * Sq;

  // prologue: TDM DMA of first K/V tiles into buffer 0 (wave 0 issues)
  if (wv == 0) {
    tdm_load_2d((unsigned)(uintptr_t)&Kt[0][0][0], Kg, DKd, Sq, DKd, DKd, 32);
    tdm_load_2d((unsigned)(uintptr_t)&Vs[0][0][0], Vg, Sq, DKd, Sq, 32, DKd);
  }

  for (int kv = 0; kv < Sq; kv += 32) {
    const int cur = (kv >> 5) & 1;
    if (wv == 0) __builtin_amdgcn_s_wait_tensorcnt(0);
    __syncthreads();                       // tile `cur` visible to all waves
    if (wv == 0 && (kv + 32) < Sq) {       // prefetch next tile into other buf
      tdm_load_2d((unsigned)(uintptr_t)&Kt[cur ^ 1][0][0],
                  Kg + (size_t)(kv + 32) * DKd, DKd, Sq, DKd, DKd, 32);
      tdm_load_2d((unsigned)(uintptr_t)&Vs[cur ^ 1][0][0],
                  Vg + (kv + 32), Sq, DKd, Sq, 32, DKd);
    }

    // scores: two 16x16 tiles (keys 0-15, 16-31), K-dim = DK = 2x32
    AccF c0, c1;
#pragma unroll
    for (int i = 0; i < 8; ++i) { c0.f[i] = 0.f; c1.f[i] = 0.f; }
    {
      FragB16 kb;
      const uint4* k0 = reinterpret_cast<const uint4*>(&Kt[cur][ln][0]);
      const uint4* k1 = reinterpret_cast<const uint4*>(&Kt[cur][16 + ln][0]);
      kb.q[0] = k0[2 * hl];     kb.q[1] = k0[2 * hl + 1];
      c0.v = wmma_bf16(qa0.v, kb.v, c0.v);
      kb.q[0] = k0[4 + 2 * hl]; kb.q[1] = k0[4 + 2 * hl + 1];
      c0.v = wmma_bf16(qa1.v, kb.v, c0.v);
      kb.q[0] = k1[2 * hl];     kb.q[1] = k1[2 * hl + 1];
      c1.v = wmma_bf16(qa0.v, kb.v, c1.v);
      kb.q[0] = k1[4 + 2 * hl]; kb.q[1] = k1[4 + 2 * hl + 1];
      c1.v = wmma_bf16(qa1.v, kb.v, c1.v);
    }

    // mask + scale; online softmax with wave-common running max
    const int mk0 = mg[kv + ln];
    const int mk1 = mg[kv + 16 + ln];
    float s0[8], s1[8];
    float lm = -1e30f;
#pragma unroll
    for (int i = 0; i < 8; ++i) {
      s0[i] = (mk0 != 0) ? c0.f[i] * 0.125f : -1e9f;
      s1[i] = (mk1 != 0) ? c1.f[i] * 0.125f : -1e9f;
      lm = fmaxf(lm, fmaxf(s0[i], s1[i]));
    }
    lm = fmaxf(lm, __shfl_xor(lm, 1));
    lm = fmaxf(lm, __shfl_xor(lm, 2));
    lm = fmaxf(lm, __shfl_xor(lm, 4));
    lm = fmaxf(lm, __shfl_xor(lm, 8));
    lm = fmaxf(lm, __shfl_xor(lm, 16));
    const float mn    = fmaxf(Mrun, lm);
    const float alpha = __expf(Mrun - mn);
    Mrun = mn;
#pragma unroll
    for (int i = 0; i < 8; ++i) {
      float p0 = __expf(s0[i] - mn);
      float p1 = __expf(s1[i] - mn);
      const int m = i + 8 * hl;
      Pb[wv][m][ln]      = f2bf(p0);
      Pb[wv][m][16 + ln] = f2bf(p1);
    }
#pragma unroll
    for (int t4 = 0; t4 < 4; ++t4)
#pragma unroll
      for (int i = 0; i < 8; ++i) acc[t4].f[i] *= alpha;
#pragma unroll
    for (int i = 0; i < 8; ++i) lrow[i] *= alpha;

    // P as A-fragment; rowsum via WMMA(P, ones); then P·V
    FragB16 pa;
    const uint4* pr = reinterpret_cast<const uint4*>(&Pb[wv][ln][0]);
    pa.q[0] = pr[2 * hl];
    pa.q[1] = pr[2 * hl + 1];
    AccF zl, ls;
#pragma unroll
    for (int i = 0; i < 8; ++i) zl.f[i] = 0.f;
    ls.v = wmma_bf16(pa.v, ones.v, zl.v);
#pragma unroll
    for (int i = 0; i < 8; ++i) lrow[i] += ls.f[i];
#pragma unroll
    for (int t4 = 0; t4 < 4; ++t4) {
      FragB16 vb;
      const uint4* vr = reinterpret_cast<const uint4*>(&Vs[cur][t4 * 16 + ln][0]);
      vb.q[0] = vr[2 * hl];
      vb.q[1] = vr[2 * hl + 1];
      acc[t4].v = wmma_bf16(pa.v, vb.v, acc[t4].v);
    }
    __syncthreads();                       // all done with buffer `cur`
  }

  float inv[8];
#pragma unroll
  for (int i = 0; i < 8; ++i) inv[i] = 1.0f / lrow[i];
  const int h = bh % Hh;
#pragma unroll
  for (int t4 = 0; t4 < 4; ++t4)
#pragma unroll
    for (int i = 0; i < 8; ++i) {
      int srow = qt * 128 + wv * 16 + i + 8 * hl;
      int col  = h * DKd + t4 * 16 + ln;
      O[(size_t)(b * Sq + srow) * Dm + col] = f2bf(acc[t4].f[i] * inv[i]);
    }
}

// ---------------------------------------------------------------------------
extern "C" void kernel_launch(void* const* d_in, const int* in_sizes, int n_in,
                              void* d_out, int out_size, void* d_ws, size_t ws_size,
                              hipStream_t stream) {
  const float* q   = (const float*)d_in[0];
  const float* k   = (const float*)d_in[1];
  const float* v   = (const float*)d_in[2];
  const int*   msk = (const int*)d_in[3];
  const float* w_q = (const float*)d_in[4];
  const float* b_q = (const float*)d_in[5];
  const float* w_k = (const float*)d_in[6];
  const float* b_k = (const float*)d_in[7];
  const float* w_v = (const float*)d_in[8];
  const float* b_v = (const float*)d_in[9];
  const float* w_o = (const float*)d_in[10];
  const float* b_o = (const float*)d_in[11];

  char* ws = (char*)d_ws;
  const size_t WT  = (size_t)Dm * Dm * 2;
  const size_t QKV = (size_t)Bn * Hh * Sq * DKd * 2;
  unsigned short* Wt_q = (unsigned short*)(ws);
  unsigned short* Wt_k = (unsigned short*)(ws + WT);
  unsigned short* Wt_v = (unsigned short*)(ws + 2 * WT);
  unsigned short* Wt_o = (unsigned short*)(ws + 3 * WT);
  unsigned short* Qbf  = (unsigned short*)(ws + 4 * WT);
  unsigned short* Kbf  = (unsigned short*)(ws + 4 * WT + QKV);
  unsigned short* Vtb  = (unsigned short*)(ws + 4 * WT + 2 * QKV);
  unsigned short* Obf  = (unsigned short*)(ws + 4 * WT + 3 * QKV);

  dim3 tb(256);
  const int wblocks = (Dm * Dm) / 256;
  wtrans_kernel<<<wblocks, tb, 0, stream>>>(w_q, Wt_q);
  wtrans_kernel<<<wblocks, tb, 0, stream>>>(w_k, Wt_k);
  wtrans_kernel<<<wblocks, tb, 0, stream>>>(w_v, Wt_v);
  wtrans_kernel<<<wblocks, tb, 0, stream>>>(w_o, Wt_o);

  dim3 gg(Dm / 64, (Bn * Sq) / 128);
  gemm_kernel<false, 0><<<gg, tb, 0, stream>>>(q, Wt_q, b_q, Qbf);
  gemm_kernel<false, 0><<<gg, tb, 0, stream>>>(k, Wt_k, b_k, Kbf);
  gemm_kernel<false, 1><<<gg, tb, 0, stream>>>(v, Wt_v, b_v, Vtb);

  dim3 ga(Sq / 128, Bn * Hh);
  attn_kernel<<<ga, tb, 0, stream>>>(Qbf, Kbf, Vtb, msk, Obf);

  gemm_kernel<true, 2><<<gg, tb, 0, stream>>>(Obf, Wt_o, b_o, d_out);

  (void)in_sizes; (void)n_in; (void)out_size; (void)ws_size;
}
